// LaserAttention_10144712753262
// MI455X (gfx1250) — matrix operand order repaired and checked by reading, not compiled, not run on hardware
//
#include <hip/hip_runtime.h>

#define B_ 2
#define L_ 4096
#define S_ 4096
#define H_ 8
#define E_ 64
#define D_ 64

// LDS strides in 32-bit words (odd strides -> conflict-free fragment reads)
#define KSTRIDE 35
#define VSTRIDE 35
#define PSTRIDE 33

typedef __attribute__((ext_vector_type(16))) __bf16 v16bf;
typedef __attribute__((ext_vector_type(2)))  __bf16 v2bf;
typedef __attribute__((ext_vector_type(8)))  float  v8f;

union FragAB { v16bf v; unsigned int u[8]; };
union FragC  { v8f  v; float f[8]; };

// native f32->bf16 (RNE); pair form should match v_cvt_pk_bf16_f32
__device__ __forceinline__ unsigned pack_bf16(float lo, float hi) {
  v2bf p;
  p.x = (__bf16)lo;
  p.y = (__bf16)hi;
  return __builtin_bit_cast(unsigned, p);
}
__device__ __forceinline__ unsigned short f2bf(float f) {
  return __builtin_bit_cast(unsigned short, (__bf16)f);
}

// ---------------------------------------------------------------------------
// Kernel 1: value_max[b,h,d] = max over S of values[b,s,h,d]
// ---------------------------------------------------------------------------
__global__ __launch_bounds__(512) void vmax_kernel(const float* __restrict__ V,
                                                   float* __restrict__ vmax) {
  const int bh = blockIdx.x;           // 0..B*H-1
  const int b = bh / H_, h = bh % H_;
  const int t = threadIdx.x;           // 0..511
  const int d = t & 63;
  const int slice = t >> 6;            // 0..7, each covers S/8 = 512 rows

  const float* p = V + (size_t)b * S_ * H_ * D_ + (size_t)h * D_ + d
                     + (size_t)slice * 512 * (H_ * D_);
  float m = -3.0e38f;
  for (int s = 0; s < 512; ++s) m = fmaxf(m, p[(size_t)s * (H_ * D_)]);

  __shared__ float red[8][64];
  red[slice][d] = m;
  __syncthreads();
  if (t < 64) {
    float mm = red[0][t];
#pragma unroll
    for (int i = 1; i < 8; ++i) mm = fmaxf(mm, red[i][t]);
    vmax[bh * 64 + t] = mm;
  }
}

// ---------------------------------------------------------------------------
// Kernel 2: flash attention with log-domain V, bf16 WMMA, online softmax.
// Block = 64 Q rows (4 waves x 16 rows), loops S in 64-key chunks.
// ---------------------------------------------------------------------------
__global__ __launch_bounds__(128) void laser_attn_kernel(
    const float* __restrict__ Q, const float* __restrict__ K,
    const float* __restrict__ V, const float* __restrict__ vmax,
    float* __restrict__ Out) {
  __shared__ unsigned ktile[64 * KSTRIDE];      // [key][e/2] bf16 pairs
  __shared__ unsigned vtile[64 * VSTRIDE];      // [d][s/2]  bf16 pairs (exp'd, transposed)
  __shared__ unsigned ptile[4][16 * PSTRIDE];   // per-wave P relayout scratch

  const int t    = threadIdx.x;
  const int lane = t & 31;
  const int w    = t >> 5;        // wave id 0..3
  const int g    = lane >> 4;     // lane-half
  const int n    = lane & 15;

  const int b = blockIdx.z, h = blockIdx.y;
  const int qbase = blockIdx.x * 64 + w * 16;

  // fold softmax scale and log2(e) into Q so scores are in exp2 domain
  const float qscale = 0.125f * 1.44269504088896340736f;

  // --- Q A-fragments (row = lane&15; K halves split by lane-half, contiguous k = 16*g + s) ---
  const float* qrow = Q + (((size_t)b * L_ + qbase + n) * H_ + h) * E_;
  FragAB qf0, qf1;
#pragma unroll
  for (int i = 0; i < 8; ++i) {
    const int k = 2 * i + 16 * g;
    float2 x0 = *(const float2*)(qrow + k);
    float2 x1 = *(const float2*)(qrow + k + 32);
    qf0.u[i] = pack_bf16(x0.x * qscale, x0.y * qscale);
    qf1.u[i] = pack_bf16(x1.x * qscale, x1.y * qscale);
  }

  const float* kbase = K + ((size_t)b * S_ * H_ + h) * E_;
  const float* vbase = V + ((size_t)b * S_ * H_ + h) * D_;
  const float* vmrow = vmax + (b * H_ + h) * D_;

  // thread-constant staging coordinates (hoisted out of the S loop)
  const int k_e2   = t & 31;        // K staging: e-pair index, constant
  const int k_key0 = t >> 5;        // K staging: key = k_key0 + 4*i
  const int v_d    = t & 63;        // V staging: d, constant per thread
  const int v_s20  = t >> 6;        // V staging: s2 = v_s20 + 2*i
  const float vm_stage = vmrow[v_d];  // loop-invariant vmax for V staging

  FragC o[4];
  float m_r[8], l_r[8];
#pragma unroll
  for (int j = 0; j < 4; ++j)
#pragma unroll
    for (int r = 0; r < 8; ++r) o[j].f[r] = 0.f;
#pragma unroll
  for (int r = 0; r < 8; ++r) { m_r[r] = -3.0e38f; l_r[r] = 0.f; }

  for (int s0 = 0; s0 < S_; s0 += 64) {
    __syncthreads();   // previous chunk's LDS reads complete before restaging

    // ---- stage K chunk: [key][e] bf16 pairs (cvt once here) ----
#pragma unroll
    for (int i = 0; i < 16; ++i) {
      const int key = k_key0 + 4 * i;
      float2 kv = *(const float2*)(kbase + (size_t)(s0 + key) * (H_ * E_) + 2 * k_e2);
      ktile[key * KSTRIDE + k_e2] = pack_bf16(kv.x, kv.y);
    }
    // ---- stage V chunk transposed + exp(v - vmax): vtile[d][s2] ----
#pragma unroll
    for (int i = 0; i < 16; ++i) {
      const int s2 = v_s20 + 2 * i;
      float v0 = vbase[(size_t)(s0 + 2 * s2) * (H_ * D_) + v_d];
      float v1 = vbase[(size_t)(s0 + 2 * s2 + 1) * (H_ * D_) + v_d];
      float e0 = __builtin_amdgcn_exp2f((v0 - vm_stage) * 1.44269504f);
      float e1 = __builtin_amdgcn_exp2f((v1 - vm_stage) * 1.44269504f);
      vtile[v_d * VSTRIDE + s2] = pack_bf16(e0, e1);
    }
    __syncthreads();

    // ---- GEMM1: S = Q * K^T (exp2 domain), 4 key subtiles x (2 WMMA, K=64) ----
    FragC sc[4];
#pragma unroll
    for (int j = 0; j < 4; ++j) {
      FragAB b0, b1;
#pragma unroll
      for (int i = 0; i < 8; ++i) {
        const int row = j * 16 + n;                       // key column of B
        b0.u[i] = ktile[row * KSTRIDE + i + 8 * g];       // e = 2i+16g, 2i+16g+1
        b1.u[i] = ktile[row * KSTRIDE + i + 8 * g + 16];  // +32 in e
      }
      v8f c = {};
      c = __builtin_amdgcn_wmma_f32_16x16x32_bf16(false, qf0.v, false, b0.v,
                                                  (short)0, c, false, false);
      c = __builtin_amdgcn_wmma_f32_16x16x32_bf16(false, qf1.v, false, b1.v,
                                                  (short)0, c, false, false);
      sc[j].v = c;
    }

    // ---- online softmax (per-row stats: row = 8*g + r, 16-lane half reduce) ----
#pragma unroll
    for (int r = 0; r < 8; ++r) {
      float loc = fmaxf(fmaxf(sc[0].f[r], sc[1].f[r]),
                        fmaxf(sc[2].f[r], sc[3].f[r]));
#pragma unroll
      for (int msk = 1; msk < 16; msk <<= 1)
        loc = fmaxf(loc, __shfl_xor(loc, msk, 32));
      const float m_new = fmaxf(m_r[r], loc);
      const float corr  = __builtin_amdgcn_exp2f(m_r[r] - m_new);
      m_r[r] = m_new;
      l_r[r] *= corr;
#pragma unroll
      for (int j = 0; j < 4; ++j) o[j].f[r] *= corr;
      float ps = 0.f;
#pragma unroll
      for (int j = 0; j < 4; ++j) {
        const float p = __builtin_amdgcn_exp2f(sc[j].f[r] - m_new);
        sc[j].f[r] = p;
        ps += p;
      }
#pragma unroll
      for (int msk = 1; msk < 16; msk <<= 1) ps += __shfl_xor(ps, msk, 32);
      l_r[r] += ps;
    }

    // ---- P: C-layout -> LDS (bf16) -> A-fragments (intra-wave, in-order DS) ----
    {
      unsigned short* pw = (unsigned short*)ptile[w];
#pragma unroll
      for (int r = 0; r < 8; ++r) {
        const int row = 8 * g + r;
#pragma unroll
        for (int j = 0; j < 4; ++j)
          pw[row * (2 * PSTRIDE) + j * 16 + n] = f2bf(sc[j].f[r]);
      }
    }
    FragAB a0, a1;
#pragma unroll
    for (int i = 0; i < 8; ++i) {
      const int ku = i + 8 * g;  // (k/2) per A layout
      a0.u[i] = ptile[w][n * PSTRIDE + ku];        // s in 0..31
      a1.u[i] = ptile[w][n * PSTRIDE + ku + 16];   // s in 32..63
    }

    // ---- GEMM2: O += P * exp(V), 4 d subtiles x (2 WMMA, K=64) ----
#pragma unroll
    for (int j = 0; j < 4; ++j) {
      FragAB bv0, bv1;
#pragma unroll
      for (int i = 0; i < 8; ++i) {
        const int drow = j * 16 + n;                        // d column of B
        bv0.u[i] = vtile[drow * VSTRIDE + i + 8 * g];       // s = 2i+16g, +1
        bv1.u[i] = vtile[drow * VSTRIDE + i + 8 * g + 16];  // +32 in s
      }
      v8f c = o[j].v;
      c = __builtin_amdgcn_wmma_f32_16x16x32_bf16(false, a0.v, false, bv0.v,
                                                  (short)0, c, false, false);
      c = __builtin_amdgcn_wmma_f32_16x16x32_bf16(false, a1.v, false, bv1.v,
                                                  (short)0, c, false, false);
      o[j].v = c;
    }
  }

  // ---- epilogue: out = ln(O) - ln(l) + vmax (v_log_f32 is log2) ----
  const float LN2 = 0.69314718055994530942f;
#pragma unroll
  for (int j = 0; j < 4; ++j) {
    const float vm = vmrow[j * 16 + n];
#pragma unroll
    for (int r = 0; r < 8; ++r) {
      const int q = qbase + 8 * g + r;
      const float val =
          (__builtin_amdgcn_logf(o[j].f[r]) - __builtin_amdgcn_logf(l_r[r])) * LN2 + vm;
      Out[(((size_t)b * L_ + q) * H_ + h) * D_ + j * 16 + n] = val;
    }
  }
}

// ---------------------------------------------------------------------------
extern "C" void kernel_launch(void* const* d_in, const int* in_sizes, int n_in,
                              void* d_out, int out_size, void* d_ws, size_t ws_size,
                              hipStream_t stream) {
  (void)in_sizes; (void)n_in; (void)out_size; (void)ws_size;
  const float* Q = (const float*)d_in[0];   // [B,L,H,E]
  const float* K = (const float*)d_in[1];   // [B,S,H,E]
  const float* V = (const float*)d_in[2];   // [B,S,H,D]
  float* vmax = (float*)d_ws;               // [B,H,D] = 1024 floats
  float* out  = (float*)d_out;              // [B,L,H,D]

  hipLaunchKernelGGL(vmax_kernel, dim3(B_ * H_), dim3(512), 0, stream, V, vmax);
  hipLaunchKernelGGL(laser_attn_kernel, dim3(L_ / 64, H_, B_), dim3(128), 0, stream,
                     Q, K, V, vmax, out);
}
